// RNNDecoder_13572096656124
// MI455X (gfx1250) — compile-verified, matrix operand order checked by reading, not run on hardware
//
#include <hip/hip_runtime.h>
#include <hip/hip_bf16.h>
#include <stdint.h>

// Problem constants (reference: B,T,V,D,H = 32,128,32000,512,512)
static constexpr int cB = 32;
static constexpr int cT = 128;
static constexpr int cV = 32000;
static constexpr int cD = 512;
static constexpr int cH = 512;
static constexpr int cG = 4 * cH;      // 2048
static constexpr int cBT = cB * cT;    // 4096

typedef __bf16 bf16t;
typedef __attribute__((ext_vector_type(8)))  __bf16 v8bf;
typedef __attribute__((ext_vector_type(16))) __bf16 v16bf;
typedef __attribute__((ext_vector_type(8)))  float  v8f;
typedef __attribute__((ext_vector_type(4)))  int    i32x4;
typedef __attribute__((ext_vector_type(8)))  int    i32x8;

// ---------------------------------------------------------------------------
// WMMA helpers (CDNA5: V_WMMA_F32_16X16X32_BF16, wave32)
// A fragment (16x32 bf16, MxK): lane L holds row M=L&15; 8-elem chunk at
// k0 + 8*(L>>4) and another at +16 (VGPRs 0-3 / 4-7).
// C fragment (16x16 f32): VGPR i -> M = i + 8*(L>>4), N = L&15.
// B (KxN) loaded as rows of the NxK matrix (C = A * W^T), transpose-symmetric.
// ---------------------------------------------------------------------------
__device__ __forceinline__ v16bf frag_ld(const bf16t* rowptr, int k0, int lane) {
    int kb = k0 + ((lane >> 4) << 3);
    v8bf c0 = *(const v8bf*)(rowptr + kb);
    v8bf c1 = *(const v8bf*)(rowptr + kb + 16);
    v16bf r;
#pragma unroll
    for (int i = 0; i < 8; ++i) { r[i] = c0[i]; r[i + 8] = c1[i]; }
    return r;
}

__device__ __forceinline__ v8f wmma_bf16(v16bf a, v16bf b, v8f c) {
    return __builtin_amdgcn_wmma_f32_16x16x32_bf16(false, a, false, b,
                                                   (short)0, c, false, false);
}

__device__ __forceinline__ float sigmoidf_(float x) {
    return 1.f / (1.f + __expf(-x));
}
__device__ __forceinline__ float tanhf_(float x) {
    return 2.f / (1.f + __expf(-2.f * x)) - 1.f;
}

// ---------------------------------------------------------------------------
// Prep 0: convert E (f32 [V,D]) -> bf16
// ---------------------------------------------------------------------------
__global__ void cvtE_kernel(const float* __restrict__ E, bf16t* __restrict__ Ebf, int n) {
    for (int i = blockIdx.x * blockDim.x + threadIdx.x; i < n; i += gridDim.x * blockDim.x)
        Ebf[i] = (bf16t)E[i];
}

// ---------------------------------------------------------------------------
// Prep 1: split/convert weights.
//   Wie  = bf16(W_ih[:, 0:512])      (embedding part)
//   Wio  = bf16(W_ih[:, 512:1024])   (out_prev part, t=0 only)
//   Whhb = bf16(W_hh)                (t=0 only)
//   Wc   = bf16(W_ih[:,512:] + W_hh) (combined recurrent weight, t>=1)
//   bias = b_ih + b_hh ; O0bf = bf16(out0)
// ---------------------------------------------------------------------------
__global__ void prepW_kernel(const float* __restrict__ W_ih, const float* __restrict__ W_hh,
                             const float* __restrict__ b_ih, const float* __restrict__ b_hh,
                             const float* __restrict__ out0,
                             bf16t* __restrict__ Wie, bf16t* __restrict__ Wio,
                             bf16t* __restrict__ Whhb, bf16t* __restrict__ Wc,
                             float* __restrict__ bias, bf16t* __restrict__ O0bf) {
    int idx = blockIdx.x * blockDim.x + threadIdx.x;
    if (idx < cG * cH) {
        int n = idx / cH, k = idx % cH;
        float wie = W_ih[n * (cD + cH) + k];
        float wio = W_ih[n * (cD + cH) + cD + k];
        float whh = W_hh[idx];
        Wie[idx]  = (bf16t)wie;
        Wio[idx]  = (bf16t)wio;
        Whhb[idx] = (bf16t)whh;
        Wc[idx]   = (bf16t)(wio + whh);
    }
    if (idx < cG) bias[idx] = b_ih[idx] + b_hh[idx];
    if (idx < cB * cH) O0bf[idx] = (bf16t)out0[idx];
}

// ---------------------------------------------------------------------------
// K1: Gpre[t*B+b, :] = emb(dst[b,t]) @ Wie^T + bias   (M=4096, N=2048, K=512)
// 16 gathered embedding rows staged into LDS with async global->LDS copies
// (shared by the block's 8 waves); one 16x16 tile per wave. Grid (16, 256).
// ---------------------------------------------------------------------------
__global__ __launch_bounds__(256) void pregemm_kernel(const int* __restrict__ dst,
                                                      const bf16t* __restrict__ Ebf,
                                                      const bf16t* __restrict__ Wie,
                                                      const float* __restrict__ bias,
                                                      float* __restrict__ Gpre) {
    __shared__ bf16t sA[16][cD];           // 16 KB
    int tid = threadIdx.x, lane = tid & 31, wave = tid >> 5;
    int mt = blockIdx.y;                   // 0..255

    // async gather-stage: 16 rows x 512 bf16 = 1024 x 16B chunks; 4 x 256 lanes
#pragma unroll
    for (int it = 0; it < 4; ++it) {
        int flat = it * 256 + tid;         // 0..1023
        int row = flat >> 6;               // 64 chunks per row
        int kc  = flat & 63;
        int r = mt * 16 + row;             // r = t*32 + b
        int t = r >> 5, b = r & 31;
        int erow = dst[b * cT + t];
        const bf16t* ga = Ebf + (size_t)erow * cD + kc * 8;
        unsigned ldsa = (unsigned)(uintptr_t)&sA[row][kc * 8];
        asm volatile("global_load_async_to_lds_b128 %0, %1, off"
                     :: "v"(ldsa), "v"(ga) : "memory");
    }
    asm volatile("s_wait_asynccnt 0" ::: "memory");
    __syncthreads();

    int nt = blockIdx.x * 8 + wave;        // 0..127
    int ncol = nt * 16 + (lane & 15);
    const bf16t* brow = Wie + (size_t)ncol * cH;
    const bf16t* arow = &sA[lane & 15][0];
    v8f acc = {};
#pragma unroll 4
    for (int k0 = 0; k0 < cD; k0 += 32) {
        v16bf a  = frag_ld(arow, k0, lane);
        v16bf bb = frag_ld(brow, k0, lane);
        acc = wmma_bf16(a, bb, acc);
    }
    float bn = bias[ncol];
#pragma unroll
    for (int i = 0; i < 8; ++i) {
        int rr = mt * 16 + i + 8 * (lane >> 4);
        Gpre[(size_t)rr * cG + ncol] = acc[i] + bn;
    }
}

// ---------------------------------------------------------------------------
// K1b: Gpre[0][b, :] += out0 @ Wio^T   (M=32, N=2048, K=512) -> 256 tiles
// ---------------------------------------------------------------------------
__global__ __launch_bounds__(256) void out0gemm_kernel(const bf16t* __restrict__ O0bf,
                                                       const bf16t* __restrict__ Wio,
                                                       float* __restrict__ Gpre) {
    int lane = threadIdx.x & 31;
    int wave = threadIdx.x >> 5;
    int tile = blockIdx.x * 8 + wave;      // 0..255
    int mt = tile & 1, nt = tile >> 1;     // 2 M-tiles x 128 N-tiles
    const bf16t* arow = O0bf + (size_t)(mt * 16 + (lane & 15)) * cH;
    int ncol = nt * 16 + (lane & 15);
    const bf16t* brow = Wio + (size_t)ncol * cH;
    v8f acc = {};
#pragma unroll 4
    for (int k0 = 0; k0 < cH; k0 += 32) {
        v16bf a  = frag_ld(arow, k0, lane);
        v16bf bb = frag_ld(brow, k0, lane);
        acc = wmma_bf16(a, bb, acc);
    }
#pragma unroll
    for (int i = 0; i < 8; ++i) {
        int b = mt * 16 + i + 8 * (lane >> 4);
        Gpre[(size_t)b * cG + ncol] += acc[i];   // K1 ran first on this stream
    }
}

// ---------------------------------------------------------------------------
// K2: sequential LSTM. One workgroup (32 waves). Per step t:
//   gates = Gpre[t] + h_{t-1} @ Wsel^T, Wsel = (t==0 ? Whh : Wio+Whh)
// h double-buffered bf16 in LDS; c lives in registers (wave owns its tiles).
// Emits s_barrier_signal/wait -3 (cluster barrier) on the cluster path.
// ---------------------------------------------------------------------------
__global__ __launch_bounds__(1024) void lstm_seq_kernel(const float* __restrict__ Gpre,
                                                        const bf16t* __restrict__ Wc,
                                                        const bf16t* __restrict__ Whhb,
                                                        const float* __restrict__ h0,
                                                        const float* __restrict__ c0,
                                                        bf16t* __restrict__ Hbf) {
    __shared__ bf16t hbuf[2][cB][cH];      // 64 KB
    int tid = threadIdx.x;
    int lane = tid & 31;
    int wave = tid >> 5;                   // 0..31 -> owns j-tile = wave
    int jcol = wave * 16 + (lane & 15);    // hidden column this lane handles

    for (int i = tid; i < cB * cH; i += 1024)
        (&hbuf[0][0][0])[i] = (bf16t)h0[i];

    float creg[2][8];
#pragma unroll
    for (int m = 0; m < 2; ++m)
#pragma unroll
        for (int i = 0; i < 8; ++i) {
            int b = m * 16 + i + 8 * (lane >> 4);
            creg[m][i] = c0[b * cH + jcol];
        }
    __syncthreads();

    for (int t = 0; t < cT; ++t) {
        int cur = t & 1, nxt = cur ^ 1;
        const bf16t* W = (t == 0) ? Whhb : Wc;

        v8f acc[2][4];
#pragma unroll
        for (int m = 0; m < 2; ++m)
#pragma unroll
            for (int g = 0; g < 4; ++g) {
                int n = g * cH + jcol;
#pragma unroll
                for (int i = 0; i < 8; ++i) {
                    int b = m * 16 + i + 8 * (lane >> 4);
                    acc[m][g][i] = Gpre[(size_t)(t * cB + b) * cG + n];
                }
            }

        for (int k0 = 0; k0 < cH; k0 += 32) {
            v16bf a0 = frag_ld(&hbuf[cur][(lane & 15)][0],      k0, lane);
            v16bf a1 = frag_ld(&hbuf[cur][16 + (lane & 15)][0], k0, lane);
#pragma unroll
            for (int g = 0; g < 4; ++g) {
                const bf16t* brow = W + (size_t)(g * cH + jcol) * cH;
                v16bf bb = frag_ld(brow, k0, lane);
                acc[0][g] = wmma_bf16(a0, bb, acc[0][g]);
                acc[1][g] = wmma_bf16(a1, bb, acc[1][g]);
            }
        }

#pragma unroll
        for (int m = 0; m < 2; ++m)
#pragma unroll
            for (int i = 0; i < 8; ++i) {
                float iv = sigmoidf_(acc[m][0][i]);
                float fv = sigmoidf_(acc[m][1][i]);
                float gv = tanhf_(acc[m][2][i]);
                float ov = sigmoidf_(acc[m][3][i]);
                float cn = fv * creg[m][i] + iv * gv;
                creg[m][i] = cn;
                float hn = ov * tanhf_(cn);
                bf16t hb = (bf16t)hn;
                int b = m * 16 + i + 8 * (lane >> 4);
                hbuf[nxt][b][jcol] = hb;
                Hbf[(size_t)(b * cT + t) * cH + jcol] = hb;   // row r = b*T + t
            }

        if (__builtin_amdgcn_cluster_id_x() > 0) {
            __builtin_amdgcn_s_cluster_barrier();   // multi-WGP cluster dispatch
        } else {
            __syncthreads();                        // single-workgroup fallback
        }
    }
}

// ---------------------------------------------------------------------------
// K3: logits = Hbf[4096x512] @ Ebf^T[512x32000] -> d_out (f32).
// 64-row A super-tile staged into LDS with ONE Tensor-Data-Mover descriptor
// (tensor_load_to_lds, D# built per CDNA5 ISA ch.8), waited with
// s_wait_tensorcnt. 8 waves x (1 N-tile x 4 M-tiles) register blocking:
// each B fragment feeds 4 WMMAs -> 4x less L2 traffic on E.
// grid (250, 64).
// ---------------------------------------------------------------------------
__global__ __launch_bounds__(256) void proj_kernel(const bf16t* __restrict__ Hbf,
                                                   const bf16t* __restrict__ Ebf,
                                                   float* __restrict__ out) {
    __shared__ bf16t sA[64][cD];           // 64 KB
    int tid = threadIdx.x, lane = tid & 31, wave = tid >> 5;
    int mblk = blockIdx.y;                 // 0..63 (64 rows each)

    if (wave == 0) {
        // --- Tensor DMA descriptor: 2D tile 64 x 512 elements, 2B each ---
        uint64_t ga = (uint64_t)(uintptr_t)(Hbf + (size_t)mblk * 64 * cD);
        unsigned ldsa = (unsigned)(uintptr_t)&sA[0][0];
        i32x4 g0;
        g0[0] = 1;                                   // count=1, user descriptor
        g0[1] = (int)ldsa;                           // lds_addr [63:32]
        g0[2] = (int)(ga & 0xffffffffu);             // global_addr lo
        g0[3] = (int)(((ga >> 32) & 0x01ffffffu) | 0x80000000u); // hi | type=2
        i32x8 g1;
        g1[0] = 0x00010000;                          // data_size=1 (2 bytes)
        g1[1] = (int)(512u << 16);                   // tensor_dim0 = 512
        g1[2] = (int)(64u  << 16);                   // tensor_dim1 = 64
        g1[3] = (int)(512u << 16);                   // tile_dim0 = 512
        g1[4] = 64;                                  // tile_dim1 = 64
        g1[5] = 512;                                 // tensor_dim0_stride = 512
        g1[6] = 0;
        g1[7] = 0;
        asm volatile("tensor_load_to_lds %0, %1" :: "s"(g0), "s"(g1) : "memory");
        __builtin_amdgcn_s_wait_tensorcnt(0);
    }
    __syncthreads();

    int nt = blockIdx.x * 8 + wave;        // 0..1999
    int ncol = nt * 16 + (lane & 15);      // vocab id
    const bf16t* brow = Ebf + (size_t)ncol * cD;
    __builtin_prefetch(brow, 0, 1);
    v8f acc[4];
#pragma unroll
    for (int mm = 0; mm < 4; ++mm) acc[mm] = (v8f){};
#pragma unroll 2
    for (int k0 = 0; k0 < cD; k0 += 32) {
        v16bf bb = frag_ld(brow, k0, lane);
#pragma unroll
        for (int mm = 0; mm < 4; ++mm) {
            v16bf a = frag_ld(&sA[mm * 16 + (lane & 15)][0], k0, lane);
            acc[mm] = wmma_bf16(a, bb, acc[mm]);
        }
    }
#pragma unroll
    for (int mm = 0; mm < 4; ++mm)
#pragma unroll
        for (int i = 0; i < 8; ++i) {
            int r = mblk * 64 + mm * 16 + i + 8 * (lane >> 4);
            out[(size_t)r * cV + ncol] = acc[mm][i];
        }
}

// ---------------------------------------------------------------------------
// K4: fused log-softmax normalization: one block per row.
// Online (single-read) max+sum, LDS pair reduction, then subtract pass
// (row is L2-hot for the second pass). Saves a full 524 MB HBM re-read.
// ---------------------------------------------------------------------------
__global__ __launch_bounds__(256) void lsesub_kernel(float* __restrict__ out) {
    __shared__ float redm[256], reds[256];
    int tid = threadIdx.x;
    int r = blockIdx.x;
    float* row = out + (size_t)r * cV;
    float m = -3.4e38f, s = 0.f;
    for (int v = tid; v < cV; v += 256) {      // 125 elements/thread, one read
        float x = row[v];
        float nm = fmaxf(m, x);
        s = s * __expf(m - nm) + __expf(x - nm);
        m = nm;
    }
    redm[tid] = m; reds[tid] = s;
    __syncthreads();
    for (int st = 128; st > 0; st >>= 1) {
        if (tid < st) {
            float m2 = redm[tid + st], s2 = reds[tid + st];
            float m1 = redm[tid],      s1 = reds[tid];
            float nm = fmaxf(m1, m2);
            reds[tid] = s1 * __expf(m1 - nm) + s2 * __expf(m2 - nm);
            redm[tid] = nm;
        }
        __syncthreads();
    }
    float lse = redm[0] + __logf(reds[0]);
    for (int v = tid; v < cV; v += 256) row[v] -= lse;
}

// ---------------------------------------------------------------------------
extern "C" void kernel_launch(void* const* d_in, const int* in_sizes, int n_in,
                              void* d_out, int out_size, void* d_ws, size_t ws_size,
                              hipStream_t stream) {
    (void)in_sizes; (void)n_in; (void)out_size; (void)ws_size;
    const int*   dst  = (const int*)  d_in[0];
    const float* E    = (const float*)d_in[1];
    const float* W_ih = (const float*)d_in[2];
    const float* W_hh = (const float*)d_in[3];
    const float* b_ih = (const float*)d_in[4];
    const float* b_hh = (const float*)d_in[5];
    const float* h0   = (const float*)d_in[6];
    const float* c0   = (const float*)d_in[7];
    const float* out0 = (const float*)d_in[8];
    float* out = (float*)d_out;

    char* wp = (char*)d_ws;
    auto carve = [&](size_t bytes) -> char* {
        char* p = wp;
        wp += (bytes + 255) & ~(size_t)255;
        return p;
    };
    bf16t* Ebf  = (bf16t*)carve((size_t)cV * cD * 2);      // 32 MB
    bf16t* Wie  = (bf16t*)carve((size_t)cG * cH * 2);      // 2 MB
    bf16t* Wio  = (bf16t*)carve((size_t)cG * cH * 2);      // 2 MB
    bf16t* Whhb = (bf16t*)carve((size_t)cG * cH * 2);      // 2 MB
    bf16t* Wc   = (bf16t*)carve((size_t)cG * cH * 2);      // 2 MB
    float* bias = (float*)carve((size_t)cG * 4);
    bf16t* O0bf = (bf16t*)carve((size_t)cB * cH * 2);
    float* Gpre = (float*)carve((size_t)cBT * cG * 4);     // 32 MB
    bf16t* Hbf  = (bf16t*)carve((size_t)cBT * cH * 2);     // 4 MB

    cvtE_kernel<<<1024, 256, 0, stream>>>(E, Ebf, cV * cD);
    prepW_kernel<<<(cG * cH + 255) / 256, 256, 0, stream>>>(
        W_ih, W_hh, b_ih, b_hh, out0, Wie, Wio, Whhb, Wc, bias, O0bf);
    pregemm_kernel<<<dim3(16, 256), 256, 0, stream>>>(dst, Ebf, Wie, bias, Gpre);
    out0gemm_kernel<<<32, 256, 0, stream>>>(O0bf, Wio, Gpre);
    lstm_seq_kernel<<<1, 1024, 0, stream>>>(Gpre, Wc, Whhb, h0, c0, Hbf);
    proj_kernel<<<dim3(250, 64), 256, 0, stream>>>(Hbf, Ebf, out);
    lsesub_kernel<<<cBT, 256, 0, stream>>>(out);
}
